// GemmaAttention_6236292514274
// MI455X (gfx1250) — compile-verified, matrix operand order checked by reading, not run on hardware
//
#include <hip/hip_runtime.h>
#include <hip/hip_bf16.h>

// ---------------- problem constants ----------------
constexpr int cB   = 2;
constexpr int cS   = 2048;
constexpr int cHID = 2048;
constexpr int cH   = 8;
constexpr int cHKV = 4;
constexpr int cD   = 256;
constexpr int cSW  = 1024;
constexpr float cSOFTCAP = 50.0f;
constexpr float cSCALE   = 0.0625f;          // 256^-0.5
constexpr float cNEG     = -2.3819763e+38f;
constexpr int cNT = cB * cS;                 // 4096 tokens

// ---------------- WMMA types ----------------
typedef __attribute__((ext_vector_type(16))) __bf16 v16bf;
typedef __attribute__((ext_vector_type(8)))  float  v8f;
typedef __attribute__((ext_vector_type(4)))  int    v4i;

union FragAB { v16bf v; unsigned short u[16]; };
union FragC  { v8f   v; float f[8]; };

__device__ __forceinline__ unsigned short f2bf(float x) {
    unsigned int u = __float_as_uint(x);
    u += 0x7FFFu + ((u >> 16) & 1u);          // round to nearest even
    return (unsigned short)(u >> 16);
}

// hardware tanh on gfx1250 if the builtin exists; safe __expf fallback else
__device__ __forceinline__ float fast_tanh(float x) {
#if __has_builtin(__builtin_amdgcn_tanhf)
    return __builtin_amdgcn_tanhf(x);
#else
    float s = (x < 0.0f) ? -1.0f : 1.0f;
    float t = __expf(-2.0f * fabsf(x));       // in (0,1]: no overflow
    return s * (1.0f - t) / (1.0f + t);
#endif
}

// 16-byte global -> LDS copy. CDNA5 async-to-LDS path (ASYNCcnt, no VGPR
// round trip); plain b128 load/ds_store fallback.
#if __has_builtin(__builtin_amdgcn_global_load_async_to_lds_b128) && \
    __has_builtin(__builtin_amdgcn_s_wait_asynccnt)
#define ASYNC_LDS_OK 1
#endif

__device__ __forceinline__ void cp16(const unsigned short* g, unsigned short* l) {
#ifdef ASYNC_LDS_OK
    __builtin_amdgcn_global_load_async_to_lds_b128(
        (__attribute__((address_space(1))) v4i*)(v4i*)(void*)g,
        (__attribute__((address_space(3))) v4i*)(v4i*)(void*)l, 0, 0);
#else
    *(uint4*)l = *(const uint4*)g;
#endif
}
__device__ __forceinline__ void cp_wait() {
#ifdef ASYNC_LDS_OK
    __builtin_amdgcn_s_wait_asynccnt(0);
#endif
}

// Load a 16x32 bf16 fragment (A layout; row-major B^T uses the same pattern).
// Per ISA 7.12.2: lane l -> row (l&15), element j -> col
// (j>>3)*16 + (j&7) + (l>>4)*8.
__device__ __forceinline__ v16bf load_frag(const unsigned short* base, int row0, int stride) {
    int lane = threadIdx.x & 31;
    const unsigned short* p = base + (unsigned)(row0 + (lane & 15)) * (unsigned)stride +
                              ((lane >> 4) << 3);
    FragAB fr;
#pragma unroll
    for (int j = 0; j < 16; j += 2) {
        int k = ((j >> 3) << 4) + (j & 7);
        fr.u[j]     = p[k];
        fr.u[j + 1] = p[k + 1];
    }
    return fr.v;
}

// ---------------- kernel 0: zero probs region ----------------
__global__ void zero_f32(float* p, size_t n4) {
    size_t stride = (size_t)gridDim.x * blockDim.x;
    for (size_t i = (size_t)blockIdx.x * blockDim.x + threadIdx.x; i < n4; i += stride) {
        float4 z = {0.f, 0.f, 0.f, 0.f};
        ((float4*)p)[i] = z;
    }
}

// ---------------- kernel 1: f32 -> bf16 ----------------
__global__ void cvt_bf16(const float* __restrict__ src, unsigned short* __restrict__ dst, size_t n) {
    size_t i = (size_t)blockIdx.x * blockDim.x + threadIdx.x;
    if (i < n) dst[i] = f2bf(src[i]);
}

// ---------------- kernel 2: bf16 GEMM, C = A(MxK) * B(NxK)^T ----------------
// 64x128 tile / 256-thread block; wave owns one mi and 4 ni subtiles.
// Double-buffered LDS: async copies for chunk k+1 overlap WMMA on chunk k.
constexpr int GTM = 64, GTN = 128, GK = 64;

__global__ __launch_bounds__(256) void gemm_bf16(const unsigned short* __restrict__ A,
                                                 const unsigned short* __restrict__ Bw,
                                                 float* __restrict__ C,
                                                 int M, int N, int K) {
    __shared__ unsigned short As[2][GTM * GK];   // 2 x 8 KB
    __shared__ unsigned short Bs[2][GTN * GK];   // 2 x 16 KB
    const int tm0 = blockIdx.x * GTM;
    const int tn0 = blockIdx.y * GTN;
    const int tid = threadIdx.x;
    const int wave = tid >> 5;
    const int lane = tid & 31;

    // per-thread staging slots (uint4 = 8 halfs): A 512 vecs (2/thr),
    // B 1024 vecs (4/thr)
    const unsigned short* ga[2]; unsigned la_off[2];
    const unsigned short* gb[4]; unsigned lb_off[4];
#pragma unroll
    for (int t = 0; t < 2; ++t) {
        unsigned v = (unsigned)tid + t * 256u;
        unsigned r = v >> 3, c = (v & 7u) << 3;
        ga[t] = A + (size_t)(tm0 + (int)r) * K + c;
        la_off[t] = r * GK + c;
    }
#pragma unroll
    for (int t = 0; t < 4; ++t) {
        unsigned v = (unsigned)tid + t * 256u;
        unsigned r = v >> 3, c = (v & 7u) << 3;
        gb[t] = Bw + (size_t)(tn0 + (int)r) * K + c;
        lb_off[t] = r * GK + c;
    }

    const int mi = wave & 3;
    const int nb = (wave >> 2) * 4;

    FragC acc[4];
#pragma unroll
    for (int j = 0; j < 4; ++j)
#pragma unroll
        for (int p = 0; p < 8; ++p) acc[j].f[p] = 0.0f;

    auto stage = [&](int buf, int k0) {
#pragma unroll
        for (int t = 0; t < 2; ++t) cp16(ga[t] + k0, &As[buf][la_off[t]]);
#pragma unroll
        for (int t = 0; t < 4; ++t) cp16(gb[t] + k0, &Bs[buf][lb_off[t]]);
    };

    stage(0, 0);
    cp_wait();
    __syncthreads();

    int cur = 0;
    for (int k0 = 0; k0 < K; k0 += GK) {
        if (k0 + GK < K) {
            stage(cur ^ 1, k0 + GK);                       // overlap with compute
            __builtin_prefetch(ga[0] + k0 + 2 * GK, 0, 0); // pull chunk k+2 toward L2
        }
#pragma unroll
        for (int kk = 0; kk < GK; kk += 32) {
            v16bf a = load_frag(&As[cur][kk], mi * 16, GK);
            FragAB bf[4];
#pragma unroll
            for (int j = 0; j < 4; ++j)
                bf[j].v = load_frag(&Bs[cur][kk], (nb + j) * 16, GK);
#pragma unroll
            for (int j = 0; j < 4; ++j)
                acc[j].v = __builtin_amdgcn_wmma_f32_16x16x32_bf16(
                    false, a, false, bf[j].v, (short)0, acc[j].v, false, false);
        }
        __syncthreads();   // all waves done reading buffer `cur`
        cp_wait();         // this wave's copies into the other buffer landed
        __syncthreads();   // everyone's copies landed
        cur ^= 1;
    }
#pragma unroll
    for (int j = 0; j < 4; ++j) {
        int col = tn0 + (nb + j) * 16 + (lane & 15);
#pragma unroll
        for (int p = 0; p < 8; ++p) {
            int row = tm0 + mi * 16 + p + ((lane >> 4) << 3);
            C[(size_t)row * N + col] = acc[j].f[p];
        }
    }
}

// ---------------- kernel 3: RoPE + pack to bf16 (b,h,s,d) ----------------
__global__ void rope_pack(const float* __restrict__ src, const float* __restrict__ cosb,
                          const float* __restrict__ sinb, unsigned short* __restrict__ dst,
                          int nh) {
    size_t i = (size_t)blockIdx.x * blockDim.x + threadIdx.x;
    size_t total = (size_t)cNT * nh * 128;
    if (i >= total) return;
    int d = i & 127;
    int h = (i >> 7) % nh;
    size_t t = i / ((size_t)128 * nh);
    int s = (int)(t % cS);
    int b = (int)(t / cS);
    float c = cosb[(unsigned)s * 128u + d];
    float sn = sinb[(unsigned)s * 128u + d];
    const float* row = src + t * (size_t)(nh * cD) + (size_t)h * cD;
    float x1 = row[d], x2 = row[d + 128];
    size_t o = (((size_t)b * nh + h) * cS + s) * cD;
    dst[o + d]       = f2bf(x1 * c - x2 * sn);
    dst[o + d + 128] = f2bf(x1 * sn + x2 * c);
}

// ---------------- kernel 4: pack V transposed to bf16 (b,kv,d,s) ----------
__global__ void pack_v(const float* __restrict__ vraw, unsigned short* __restrict__ Vt) {
    size_t i = (size_t)blockIdx.x * blockDim.x + threadIdx.x;
    size_t total = (size_t)cNT * cHKV * cD;
    if (i >= total) return;
    int d = i & 255;
    int kv = (i >> 8) & 3;
    size_t t = i >> 10;
    int s = (int)(t % cS);
    int b = (int)(t / cS);
    Vt[(((size_t)b * cHKV + kv) * cD + d) * cS + s] = f2bf(vraw[t * (size_t)1024 + kv * 256 + d]);
}

// ---------------- kernel 5: QK^T tile + softcap + mask -> probs (raw) -----
__global__ __launch_bounds__(256) void scores_kernel(const unsigned short* __restrict__ Qb,
                                                     const unsigned short* __restrict__ Kb,
                                                     float* __restrict__ probs) {
    const int qt = blockIdx.x;          // 0..31
    const int rel = blockIdx.y;         // 0..16
    const int bh = blockIdx.z;          // 0..15
    const int b = bh >> 3, h = bh & 7, kv = h >> 1;
    const int q0 = qt * 64;
    int kf = q0 - (cSW - 1); if (kf < 0) kf = 0;
    const int kfirst = kf >> 6;
    const int kt = kfirst + rel;
    if (kt > qt) return;
    const int k0 = kt * 64;

    __shared__ unsigned short Qs[64 * 256];
    __shared__ unsigned short Ks[64 * 256];
    const unsigned short* Qg = Qb + (((size_t)b * cH + h) * cS + q0) * cD;
    const unsigned short* Kg = Kb + (((size_t)b * cHKV + kv) * cS + k0) * cD;
    const int tid = threadIdx.x;
#pragma unroll
    for (int t = 0; t < 8; ++t) {
        unsigned i = ((unsigned)tid + t * 256u) << 3;   // half offset of uint4 slot
        cp16(Qg + i, &Qs[i]);
    }
#pragma unroll
    for (int t = 0; t < 8; ++t) {
        unsigned i = ((unsigned)tid + t * 256u) << 3;
        cp16(Kg + i, &Ks[i]);
    }
    cp_wait();
    __syncthreads();

    const int wave = tid >> 5, lane = tid & 31;
    // wave's two subtiles share mi -> one A-fragment, two B-fragments
    const int mi = wave >> 1;
    const int nb2 = (wave & 1) * 2;
    FragC acc[2];
#pragma unroll
    for (int s = 0; s < 2; ++s)
#pragma unroll
        for (int p = 0; p < 8; ++p) acc[s].f[p] = 0.0f;

#pragma unroll
    for (int ks = 0; ks < 256; ks += 32) {
        v16bf a = load_frag(Qs + ks, mi * 16, 256);
        FragAB bf[2];
#pragma unroll
        for (int s = 0; s < 2; ++s)
            bf[s].v = load_frag(Ks + ks, (nb2 + s) * 16, 256);
#pragma unroll
        for (int s = 0; s < 2; ++s)
            acc[s].v = __builtin_amdgcn_wmma_f32_16x16x32_bf16(
                false, a, false, bf[s].v, (short)0, acc[s].v, false, false);
    }

#pragma unroll
    for (int s = 0; s < 2; ++s) {
        int n = (nb2 + s) * 16 + (lane & 15);
        int k = k0 + n;
#pragma unroll
        for (int p = 0; p < 8; ++p) {
            int m = mi * 16 + p + ((lane >> 4) << 3);
            int q = q0 + m;
            float v = acc[s].f[p] * cSCALE;
            v = fast_tanh(v * (1.0f / cSOFTCAP)) * cSOFTCAP;
            if (k > q || (q - k) >= cSW) v = cNEG;
            probs[((size_t)bh * cS + q) * cS + k] = v;
        }
    }
}

// ---------------- kernel 6: softmax + P@V ----------------
__global__ __launch_bounds__(256) void softmax_pv(float* __restrict__ probs,
                                                  const unsigned short* __restrict__ Vt,
                                                  float* __restrict__ attn,
                                                  unsigned short* __restrict__ attnb) {
    const int qt = blockIdx.x;   // 0..31
    const int bh = blockIdx.y;   // 0..15
    const int b = bh >> 3, h = bh & 7, kv = h >> 1;
    const int q0 = qt * 64;
    int kf = q0 - (cSW - 1); if (kf < 0) kf = 0;
    const int kbeg = (kf >> 6) << 6;       // tile-aligned window start
    const int kend = q0 + 64;              // exclusive

    __shared__ float red[256];
    __shared__ float rmax[64];
    __shared__ float rsum[64];
    __shared__ unsigned short Ps[64 * 32];

    const int tid = threadIdx.x;
    // pass 1: row max / row sum over the window (4 threads per row)
    {
        int r = tid >> 2, g = tid & 3;
        float* row = probs + ((size_t)bh * cS + (q0 + r)) * cS;
        float m = cNEG;
        for (int c = kbeg + g; c < kend; c += 4) m = fmaxf(m, row[c]);
        red[tid] = m;
        __syncthreads();
        if (tid < 64) {
            float a = fmaxf(fmaxf(red[tid * 4], red[tid * 4 + 1]),
                            fmaxf(red[tid * 4 + 2], red[tid * 4 + 3]));
            rmax[tid] = a;
        }
        __syncthreads();
        float mm = rmax[r];
        float s = 0.0f;
        for (int c = kbeg + g; c < kend; c += 4) s += __expf(row[c] - mm);
        red[tid] = s;
        __syncthreads();
        if (tid < 64)
            rsum[tid] = red[tid * 4] + red[tid * 4 + 1] + red[tid * 4 + 2] + red[tid * 4 + 3];
        __syncthreads();
    }

    const int wave = tid >> 5, lane = tid & 31;
    const int mi = wave & 3;
    const int nbase = (wave >> 2) * 8;
    FragC acc[8];
#pragma unroll
    for (int j = 0; j < 8; ++j)
#pragma unroll
        for (int p = 0; p < 8; ++p) acc[j].f[p] = 0.0f;

    const unsigned short* Vbase = Vt + ((size_t)b * cHKV + kv) * cD * (size_t)cS;

    for (int kc = kbeg; kc < kend; kc += 32) {
        // normalize this 64x32 chunk; write final probs + bf16 staging
#pragma unroll
        for (int t = 0; t < 8; ++t) {
            int i = tid + t * 256;
            int rr = i >> 5, cc = i & 31;
            float* rp = probs + ((size_t)bh * cS + (q0 + rr)) * cS;
            float pv = __expf(rp[kc + cc] - rmax[rr]) / rsum[rr];
            rp[kc + cc] = pv;
            Ps[rr * 32 + cc] = f2bf(pv);
        }
        __syncthreads();
        v16bf a = load_frag(Ps, mi * 16, 32);
#pragma unroll
        for (int jj = 0; jj < 8; jj += 2) {     // pairs: 2 loads then 2 WMMAs
            FragAB bf[2];
#pragma unroll
            for (int s = 0; s < 2; ++s)
                bf[s].v = load_frag(Vbase + (size_t)((nbase + jj + s) * 16) * cS + kc, 0, cS);
#pragma unroll
            for (int s = 0; s < 2; ++s)
                acc[jj + s].v = __builtin_amdgcn_wmma_f32_16x16x32_bf16(
                    false, a, false, bf[s].v, (short)0, acc[jj + s].v, false, false);
        }
        __syncthreads();
    }

#pragma unroll
    for (int j = 0; j < 8; ++j) {
        int ni = nbase + j;
        int n = ni * 16 + (lane & 15);
#pragma unroll
        for (int p = 0; p < 8; ++p) {
            int m = mi * 16 + p + ((lane >> 4) << 3);
            int q = q0 + m;
            float val = acc[j].f[p];
            attn[((size_t)bh * cS + q) * cD + n] = val;
            attnb[((size_t)(b * cS + q)) * (size_t)cHID + h * cD + n] = f2bf(val);
        }
    }
}

// ---------------- host orchestration ----------------
extern "C" void kernel_launch(void* const* d_in, const int* in_sizes, int n_in,
                              void* d_out, int out_size, void* d_ws, size_t ws_size,
                              hipStream_t stream) {
    (void)in_sizes; (void)n_in; (void)out_size; (void)ws_size;
    const float* hs   = (const float*)d_in[0];
    const float* fcos = (const float*)d_in[1];
    const float* fsin = (const float*)d_in[2];
    // d_in[3] = mask (recomputed analytically on device)
    const float* wq = (const float*)d_in[4];
    const float* wk = (const float*)d_in[5];
    const float* wv = (const float*)d_in[6];
    const float* wo = (const float*)d_in[7];

    float* out   = (float*)d_out;
    float* probs = out + (size_t)cB * cS * cHID;
    float* attn  = probs + (size_t)cB * cH * cS * cS;

    char* ws = (char*)d_ws;
    size_t off = 0;
    auto alloc = [&](size_t bytes) { char* p = ws + off; off += (bytes + 255) & ~(size_t)255; return p; };
    unsigned short* hsb  = (unsigned short*)alloc((size_t)cNT * cHID * 2);
    unsigned short* wqb  = (unsigned short*)alloc((size_t)cHID * cHID * 2);
    unsigned short* wkb  = (unsigned short*)alloc((size_t)(cHKV * cD) * cHID * 2);
    unsigned short* wvb  = (unsigned short*)alloc((size_t)(cHKV * cD) * cHID * 2);
    unsigned short* wob  = (unsigned short*)alloc((size_t)cHID * cHID * 2);
    float* qraw          = (float*)alloc((size_t)cNT * (cH * cD) * 4);
    float* kraw          = (float*)alloc((size_t)cNT * (cHKV * cD) * 4);
    float* vraw          = (float*)alloc((size_t)cNT * (cHKV * cD) * 4);
    unsigned short* Qb   = (unsigned short*)alloc((size_t)cB * cH * cS * cD * 2);
    unsigned short* Kb   = (unsigned short*)alloc((size_t)cB * cHKV * cS * cD * 2);
    unsigned short* Vtb  = (unsigned short*)alloc((size_t)cB * cHKV * cD * cS * 2);
    unsigned short* attnb= (unsigned short*)alloc((size_t)cNT * cHID * 2);

    // 0) zero probs (window leaves most of it untouched)
    zero_f32<<<4096, 256, 0, stream>>>(probs, (size_t)cB * cH * cS * cS / 4);

    // 1) f32 -> bf16 conversions
    auto cvt = [&](const float* s, unsigned short* d, size_t n) {
        cvt_bf16<<<(unsigned)((n + 255) / 256), 256, 0, stream>>>(s, d, n);
    };
    cvt(hs, hsb, (size_t)cNT * cHID);
    cvt(wq, wqb, (size_t)cHID * cHID);
    cvt(wk, wkb, (size_t)(cHKV * cD) * cHID);
    cvt(wv, wvb, (size_t)(cHKV * cD) * cHID);
    cvt(wo, wob, (size_t)cHID * cHID);

    // 2) QKV projections (C = hs @ W^T)
    gemm_bf16<<<dim3(cNT / GTM, (cH * cD) / GTN), 256, 0, stream>>>(hsb, wqb, qraw, cNT, cH * cD, cHID);
    gemm_bf16<<<dim3(cNT / GTM, (cHKV * cD) / GTN), 256, 0, stream>>>(hsb, wkb, kraw, cNT, cHKV * cD, cHID);
    gemm_bf16<<<dim3(cNT / GTM, (cHKV * cD) / GTN), 256, 0, stream>>>(hsb, wvb, vraw, cNT, cHKV * cD, cHID);

    // 3) RoPE + packing
    rope_pack<<<(unsigned)(((size_t)cNT * cH * 128 + 255) / 256), 256, 0, stream>>>(qraw, fcos, fsin, Qb, cH);
    rope_pack<<<(unsigned)(((size_t)cNT * cHKV * 128 + 255) / 256), 256, 0, stream>>>(kraw, fcos, fsin, Kb, cHKV);
    pack_v<<<(unsigned)(((size_t)cNT * cHKV * cD + 255) / 256), 256, 0, stream>>>(vraw, Vtb);

    // 4) scores (raw, soft-capped, masked) into probs buffer
    scores_kernel<<<dim3(cS / 64, 17, cB * cH), 256, 0, stream>>>(Qb, Kb, probs);

    // 5) softmax + P@V -> probs (final), attn_output, bf16 attn for out-proj
    softmax_pv<<<dim3(cS / 64, cB * cH), 256, 0, stream>>>(probs, Vtb, attn, attnb);

    // 6) output projection: out = attn_flat @ wo^T
    gemm_bf16<<<dim3(cNT / GTM, cHID / GTN), 256, 0, stream>>>(attnb, wob, out, cNT, cHID, cHID);
}
